// DMN4_47124381172172
// MI455X (gfx1250) — compile-verified, matrix operand order checked by reading, not run on hardware
//
#include <hip/hip_runtime.h>
#include <hip/hip_bf16.h>
#include <math.h>

#define N_WAY  5
#define K_SHOT 5
#define B_     4
#define Q_     75
#define C_     640
#define HW     100      // M_q = h*w
#define MS     500      // K_SHOT*HW
#define MQ_PAD 112      // 7 * 16
#define NS_PAD 512      // 32 * 16
#define TEMP   2.0f
#define EPSN   1e-8f
#define NEG_INF (-3.402823e38f)

typedef __attribute__((ext_vector_type(16))) __bf16 v16bf;
typedef __attribute__((ext_vector_type(8)))  float  v8f;
typedef __attribute__((ext_vector_type(4)))  int    v4i_t;

union Frag { uint4 q[2]; v16bf v; };

__device__ __forceinline__ unsigned short f2bf(float f) {
  union { float f; unsigned u; } v; v.f = f;
  unsigned r = v.u + 0x7FFFu + ((v.u >> 16) & 1u);
  return (unsigned short)(r >> 16);
}

// --------------------------- async-to-LDS support --------------------------
#if __has_builtin(__builtin_amdgcn_global_load_async_to_lds_b128)
#define HAVE_ASYNC_LDS 1
#else
#define HAVE_ASYNC_LDS 0
#endif

__device__ __forceinline__ void copy16_to_lds(const uint4* g, uint4* l) {
#if HAVE_ASYNC_LDS
  __builtin_amdgcn_global_load_async_to_lds_b128(
      (v4i_t*)(uintptr_t)g, (v4i_t*)(uintptr_t)l, 0, 0);
#else
  *l = *g;
#endif
}

__device__ __forceinline__ void wait_async_done() {
#if HAVE_ASYNC_LDS
#if __has_builtin(__builtin_amdgcn_s_wait_asynccnt)
  __builtin_amdgcn_s_wait_asynccnt(0);
#else
  asm volatile("s_wait_asynccnt 0" ::: "memory");
#endif
#endif
}

// ------------------- scheduler steering (DS-read / WMMA) -------------------
#if __has_builtin(__builtin_amdgcn_sched_group_barrier)
#define SCHED_DS_READ(n)  __builtin_amdgcn_sched_group_barrier(0x100, (n), 0)
#define SCHED_WMMA(n)     __builtin_amdgcn_sched_group_barrier(0x008, (n), 0)
#else
#define SCHED_DS_READ(n)
#define SCHED_WMMA(n)
#endif

// ---------------------------------------------------------------------------
// Normalize query descriptors over channel dim, transpose [c][m] -> [m][c],
// convert to bf16, pad rows 100..111 with zeros.  One block per (b,q).
// ---------------------------------------------------------------------------
__global__ void qnorm_kernel(const float* __restrict__ qx,
                             unsigned short* __restrict__ Qbf) {
  int bq = blockIdx.x;                                  // 0..299
  const float* src = qx + (size_t)bq * C_ * HW;         // [c][m]
  unsigned short* dst = Qbf + (size_t)bq * MQ_PAD * C_; // [m][c]
  int lane = threadIdx.x & 31;
  int w8   = threadIdx.x >> 5;                          // 0..7
  for (int m = w8; m < MQ_PAD; m += 8) {
    if (m < HW) {
      float s = 0.f;
      for (int c = lane; c < C_; c += 32) {
        float x = src[c * HW + m];
        s += x * x;
      }
      for (int d = 16; d; d >>= 1) s += __shfl_xor(s, d, 32);
      float inv = 1.f / (sqrtf(s) + EPSN);
      for (int c = lane; c < C_; c += 32)
        dst[m * C_ + c] = f2bf(src[c * HW + m] * inv);
    } else {
      for (int c = lane; c < C_; c += 32) dst[m * C_ + c] = 0;
    }
  }
}

// ---------------------------------------------------------------------------
// Normalize support descriptors: [b, n_way*k_shot, c, hw] -> [b][w][n][c] bf16
// with n = k*HW + p padded to 512 rows.  One block per (b, w).
// ---------------------------------------------------------------------------
__global__ void snorm_kernel(const float* __restrict__ sx,
                             unsigned short* __restrict__ Sbf) {
  int bw = blockIdx.x;                                  // b*N_WAY + w
  int b = bw / N_WAY, w = bw % N_WAY;
  const float* base = sx + (size_t)b * (N_WAY * K_SHOT) * C_ * HW;
  unsigned short* dst = Sbf + (size_t)bw * NS_PAD * C_;
  int lane = threadIdx.x & 31;
  int w8   = threadIdx.x >> 5;
  for (int r = w8; r < NS_PAD; r += 8) {
    if (r < MS) {
      int k = r / HW, p = r % HW;
      const float* s0 = base + (size_t)(w * K_SHOT + k) * C_ * HW + p;
      float s = 0.f;
      for (int c = lane; c < C_; c += 32) {
        float x = s0[(size_t)c * HW];
        s += x * x;
      }
      for (int d = 16; d; d >>= 1) s += __shfl_xor(s, d, 32);
      float inv = 1.f / (sqrtf(s) + EPSN);
      for (int c = lane; c < C_; c += 32)
        dst[r * C_ + c] = f2bf(s0[(size_t)c * HW] * inv);
    } else {
      for (int c = lane; c < C_; c += 32) dst[r * C_ + c] = 0;
    }
  }
}

// ---------------------------------------------------------------------------
// Stage one 16x640 bf16 B tile (1280 uint4) into LDS, straight-line.
// ---------------------------------------------------------------------------
__device__ __forceinline__ void stage_tile(const uint4* __restrict__ g,
                                           uint4* __restrict__ l, int tid) {
  const int TILE_U4 = 16 * C_ / 8;   // 1280 = 5*224 + 160
#pragma unroll
  for (int j = 0; j < 5; ++j) copy16_to_lds(g + j * 224 + tid, l + j * 224 + tid);
  if (tid < TILE_U4 - 5 * 224) copy16_to_lds(g + 5 * 224 + tid, l + 5 * 224 + tid);
}

// ---------------------------------------------------------------------------
// Per-(b,q,class) fused cosine-similarity GEMM + per-class max/argmax.
// Grid = 300*5 blocks of 224 threads (7 waves; wave = one 16-row M tile).
// A slab register-resident (160 VGPRs); B tiles double-buffered in LDS via
// async-to-LDS; B fragments pipelined depth-2 via sched_group_barrier.
// ---------------------------------------------------------------------------
__global__ void __launch_bounds__(224)
sim_kernel(const unsigned short* __restrict__ Qbf,
           const unsigned short* __restrict__ Sbf,
           float* __restrict__ smax_out,   // [300][5][100] per-class max
           int*   __restrict__ cidx_out)   // [300][5][100] per-class argmax
{
  __shared__ unsigned short Bs[2][16 * C_];   // double-buffered B tile (40 KB)
  const int TILE_U4 = 16 * C_ / 8;

  int bqw  = blockIdx.x;      // bq*N_WAY + w
  int bq   = bqw / N_WAY;
  int w    = bqw % N_WAY;
  int b    = bq / Q_;
  int tid  = threadIdx.x;
  int lane = tid & 31;
  int wave = tid >> 5;        // 0..6 -> M tile
  int waveM = wave * 16;
  int half  = lane >> 4;      // K-half select for fragments
  int l15   = lane & 15;
  int kchunk = half * 8;

  // --- A slab (16 rows x 640) for this wave -> registers (20 frags) -------
  Frag aregs[20];
  {
    const unsigned short* Abase =
        Qbf + (size_t)bq * MQ_PAD * C_ + (size_t)(waveM + l15) * C_;
#pragma unroll
    for (int kt = 0; kt < 20; ++kt) {
      aregs[kt].q[0] = *(const uint4*)(Abase + kt * 32 + kchunk);
      aregs[kt].q[1] = *(const uint4*)(Abase + kt * 32 + kchunk + 16);
    }
  }

  float lval[8]; int lidx[8];      // per-lane running (val, idx) this class
#pragma unroll
  for (int r = 0; r < 8; ++r) { lval[r] = NEG_INF; lidx[r] = 0; }

  const uint4* classBase =
      (const uint4*)(Sbf + (size_t)(b * N_WAY + w) * NS_PAD * C_);

  stage_tile(classBase, (uint4*)&Bs[0][0], tid);

  for (int t = 0; t < 32; ++t) {
    int cur = t & 1;

    wait_async_done();
    __syncthreads();    // tile t ready in Bs[cur]; reads of Bs[1-cur] done

    if (t + 1 < 32)
      stage_tile(classBase + (size_t)(t + 1) * TILE_U4,
                 (uint4*)&Bs[1 - cur][0], tid);

    // ---- 20 WMMAs over K=640: A regs, B LDS, depth-2 pipelined frags ----
    const unsigned short* Bcur = &Bs[cur][0] + l15 * C_;
    v8f acc = {};
    Frag bfr[3];
    bfr[0].q[0] = *(const uint4*)(Bcur + kchunk);
    bfr[0].q[1] = *(const uint4*)(Bcur + kchunk + 16);
    bfr[1].q[0] = *(const uint4*)(Bcur + 32 + kchunk);
    bfr[1].q[1] = *(const uint4*)(Bcur + 32 + kchunk + 16);
#pragma unroll
    for (int kt = 0; kt < 20; ++kt) {
      if (kt + 2 < 20) {
        bfr[(kt + 2) % 3].q[0] = *(const uint4*)(Bcur + (kt + 2) * 32 + kchunk);
        bfr[(kt + 2) % 3].q[1] = *(const uint4*)(Bcur + (kt + 2) * 32 + kchunk + 16);
      }
      acc = __builtin_amdgcn_wmma_f32_16x16x32_bf16(
          false, aregs[kt].v, false, bfr[kt % 3].v, (short)0, acc, false, false);
    }
    // Force pipelined schedule: frags prefetched 2 WMMAs ahead so waits are
    // partial (s_wait_dscnt 4) instead of 0 before every WMMA.
    SCHED_DS_READ(6);
#pragma unroll
    for (int k = 0; k < 17; ++k) {
      SCHED_WMMA(1);
      SCHED_DS_READ(2);
    }
    SCHED_WMMA(3);

    // ---- branchless per-lane epilogue -----------------------------------
    int gn = t * 16 + l15;              // column within class (0..511)
    bool valid = gn < MS;
    int idx = w * MS + gn;              // merged index w*M_s + n
#pragma unroll
    for (int r = 0; r < 8; ++r) {
      float v = valid ? acc[r] : NEG_INF;
      bool tk = v > lval[r];            // strict > keeps earliest nt per lane
      lval[r] = tk ? v : lval[r];
      lidx[r] = tk ? idx : lidx[r];
    }
  }

  // ---- one cross-lane (val,idx) reduce per row; write class results -----
#pragma unroll
  for (int r = 0; r < 8; ++r) {
    float v = lval[r]; int i = lidx[r];
#pragma unroll
    for (int d = 1; d < 16; d <<= 1) {   // stays within 16-lane halves
      float ov = __shfl_xor(v, d, 32);
      int   oi = __shfl_xor(i, d, 32);
      bool tk = (ov > v) || ((ov == v) && (oi < i));   // first-idx tiebreak
      v = tk ? ov : v;
      i = tk ? oi : i;
    }
    if (l15 == 0) {
      int m = waveM + half * 8 + r;
      if (m < HW) {
        size_t o = ((size_t)bq * N_WAY + w) * HW + m;
        smax_out[o] = v;
        cidx_out[o] = i;
      }
    }
  }
}

// ---------------------------------------------------------------------------
// Fold the 5 per-class (max, argmax) into merged nearest + top-2 diff.
// ---------------------------------------------------------------------------
__global__ void combine_kernel(const float* __restrict__ smax,
                               const int* __restrict__ cidx,
                               int* __restrict__ near_out,
                               float* __restrict__ diff_out) {
  int bq = blockIdx.x;
  int m  = threadIdx.x;
  if (m < HW) {
    float gv = NEG_INF; int gi = 0;
    float t1 = NEG_INF, t2 = NEG_INF;
#pragma unroll
    for (int w = 0; w < N_WAY; ++w) {
      float v = smax[((size_t)bq * N_WAY + w) * HW + m];
      int   i = cidx[((size_t)bq * N_WAY + w) * HW + m];
      bool tg = v > gv;                 // w ascending: strict > keeps first
      gv = tg ? v : gv;
      gi = tg ? i : gi;
      float n1 = fmaxf(t1, v);
      float n2 = fmaxf(t2, fminf(t1, v));
      t1 = n1; t2 = n2;
    }
    near_out[bq * HW + m] = gi;
    diff_out[bq * HW + m] = t1 - t2;
  }
}

// ---------------------------------------------------------------------------
// Mutual-nearest mask + logits + log-softmax CE per query.
// ---------------------------------------------------------------------------
__global__ void predict_kernel(const float* __restrict__ smax,
                               const int* __restrict__ nearest,
                               const float* __restrict__ diff,
                               const int* __restrict__ query_y,
                               float* __restrict__ lossq)
{
  __shared__ float sdiff[HW];
  __shared__ int   snear[HW];
  __shared__ float ssm[N_WAY * HW];
  __shared__ float skeep[HW];
  __shared__ float spred[N_WAY];
  int bq = blockIdx.x, tid = threadIdx.x;
  if (tid < HW) {
    sdiff[tid] = diff[bq * HW + tid];
    snear[tid] = nearest[bq * HW + tid];
  }
  for (int i = tid; i < N_WAY * HW; i += blockDim.x)
    ssm[i] = smax[(size_t)bq * N_WAY * HW + i];
  __syncthreads();
  if (tid < HW) {
    int s = snear[tid];
    float bv = -1.f; int ba = 0;
    for (int mp = 0; mp < HW; ++mp) {
      float vv = (snear[mp] == s) ? sdiff[mp] : 0.f;
      bool tk = vv > bv;
      bv = tk ? vv : bv;
      ba = tk ? mp : ba;
    }
    skeep[tid] = (ba == tid) ? TEMP : 0.f;
  }
  __syncthreads();
  if (tid < N_WAY) {
    float p = 0.f;
    for (int m = 0; m < HW; ++m) p += ssm[tid * HW + m] * skeep[m];
    spred[tid] = p;
  }
  __syncthreads();
  if (tid == 0) {
    float mx = spred[0];
    for (int w = 1; w < N_WAY; ++w) mx = fmaxf(mx, spred[w]);
    float se = 0.f;
    for (int w = 0; w < N_WAY; ++w) se += expf(spred[w] - mx);
    float lse = logf(se) + mx;
    int y = query_y[bq];
    lossq[bq] = lse - spred[y];
  }
}

// ---------------------------------------------------------------------------
// Final mean reduction to scalar loss.
// ---------------------------------------------------------------------------
__global__ void reduce_kernel(const float* __restrict__ lossq,
                              float* __restrict__ out) {
  __shared__ float red[256];
  int tid = threadIdx.x;
  float s = 0.f;
  for (int i = tid; i < B_ * Q_; i += 256) s += lossq[i];
  red[tid] = s;
  __syncthreads();
  for (int st = 128; st; st >>= 1) {
    if (tid < st) red[tid] += red[tid + st];
    __syncthreads();
  }
  if (tid == 0) out[0] = red[0] / (float)(B_ * Q_);
}

// ---------------------------------------------------------------------------
extern "C" void kernel_launch(void* const* d_in, const int* in_sizes, int n_in,
                              void* d_out, int out_size, void* d_ws, size_t ws_size,
                              hipStream_t stream) {
  const float* support_xf = (const float*)d_in[0];
  // d_in[1] (support_y) is unused by the reference (class = index / K_SHOT)
  const float* query_xf   = (const float*)d_in[2];
  const int*   query_y    = (const int*)d_in[3];
  (void)in_sizes; (void)n_in; (void)out_size; (void)ws_size;

  char* ws = (char*)d_ws;
  size_t off = 0;
  auto wsalloc = [&](size_t bytes) -> void* {
    void* p = ws + off;
    off = (off + bytes + 255) & ~(size_t)255;
    return p;
  };
  unsigned short* Qbf = (unsigned short*)wsalloc((size_t)B_ * Q_ * MQ_PAD * C_ * 2);
  unsigned short* Sbf = (unsigned short*)wsalloc((size_t)B_ * N_WAY * NS_PAD * C_ * 2);
  float* smax    = (float*)wsalloc((size_t)B_ * Q_ * N_WAY * HW * 4);
  int*   cidx    = (int*)  wsalloc((size_t)B_ * Q_ * N_WAY * HW * 4);
  int*   nearest = (int*)  wsalloc((size_t)B_ * Q_ * HW * 4);
  float* diff    = (float*)wsalloc((size_t)B_ * Q_ * HW * 4);
  float* lossq   = (float*)wsalloc((size_t)B_ * Q_ * 4);

  qnorm_kernel<<<B_ * Q_, 256, 0, stream>>>(query_xf, Qbf);
  snorm_kernel<<<B_ * N_WAY, 256, 0, stream>>>(support_xf, Sbf);

  sim_kernel<<<B_ * Q_ * N_WAY, 224, 0, stream>>>(Qbf, Sbf, smax, cidx);

  combine_kernel<<<B_ * Q_, 128, 0, stream>>>(smax, cidx, nearest, diff);
  predict_kernel<<<B_ * Q_, 128, 0, stream>>>(smax, nearest, diff, query_y, lossq);
  reduce_kernel<<<1, 256, 0, stream>>>(lossq, (float*)d_out);
}
